// PN_head_22110491639922
// MI455X (gfx1250) — matrix operand order repaired and checked by reading, not compiled, not run on hardware
//
#include <hip/hip_runtime.h>
#include <hip/hip_bf16.h>

// ---------------------------------------------------------------------------
// ProtoNet head for MI455X (gfx1250, wave32).
// Heavy GEMMs: v_wmma_f32_16x16x32_bf16 with bf16x3 split precision.
// Staging:    global_load_async_to_lds_b128 (ASYNCcnt) + LDS double buffering.
// ---------------------------------------------------------------------------

typedef __attribute__((ext_vector_type(16))) __bf16 v16bf;
typedef __attribute__((ext_vector_type(8)))  float  v8f;
typedef int vi4 __attribute__((vector_size(4 * sizeof(int))));
typedef unsigned short u16;

#define Q_ROWS 8192
#define S_ROWS 4096
#define CDIM   1024
#define WAYC   64
#define KNN    5

#if __has_builtin(__builtin_amdgcn_global_load_async_to_lds_b128)
#define HAVE_ASYNC_LDS 1
#else
#define HAVE_ASYNC_LDS 0
#endif

// ----- async memory -> LDS copy (16B per lane), with synchronous fallback --
static __device__ __forceinline__ void cp16_async(const u16* g, u16* l) {
#if HAVE_ASYNC_LDS
    __builtin_amdgcn_global_load_async_to_lds_b128(
        (__attribute__((address_space(1))) vi4*)(uintptr_t)g,
        (__attribute__((address_space(3))) vi4*)(unsigned)(uintptr_t)l,
        0, 0);
#else
    *(uint4*)l = *(const uint4*)g;
#endif
}
static __device__ __forceinline__ void cp32_async(const u16* g, u16* l) {
    cp16_async(g, l);
    cp16_async(g + 8, l + 8);
}
static __device__ __forceinline__ void wait_async() {
#if HAVE_ASYNC_LDS
#if __has_builtin(__builtin_amdgcn_s_wait_asynccnt)
    __builtin_amdgcn_s_wait_asynccnt(0);
#else
    asm volatile("s_wait_asynccnt 0x0" ::: "memory");
#endif
#endif
}

// ----- float <-> bf16 helpers (bit-level, round-to-nearest-even) -----------
static __device__ __forceinline__ u16 f2bf(float f) {
    unsigned u = __float_as_uint(f);
    u += 0x7FFFu + ((u >> 16) & 1u);
    return (u16)(u >> 16);
}
static __device__ __forceinline__ float bf2f(u16 h) {
    return __uint_as_float(((unsigned)h) << 16);
}

// ---------------------------------------------------------------------------
// Kernel 1: row-wise L2 normalize (F.normalize, eps=1e-12), split bf16 hi/lo.
// One wave per row, 8 rows per 256-thread block.
// ---------------------------------------------------------------------------
__global__ __launch_bounds__(256) void l2norm_split_kernel(
    const float* __restrict__ in, int rows,
    float* __restrict__ outF, u16* __restrict__ outH, u16* __restrict__ outL)
{
    int w    = threadIdx.x >> 5;
    int lane = threadIdx.x & 31;
    int row  = blockIdx.x * 8 + w;
    if (row >= rows) return;

    const float* x = in + (size_t)row * CDIM;
    float ss = 0.0f;
    for (int k = lane; k < CDIM; k += 32) { float v = x[k]; ss += v * v; }
#pragma unroll
    for (int off = 16; off > 0; off >>= 1) ss += __shfl_xor(ss, off, 32);

    float inv = 1.0f / fmaxf(sqrtf(ss), 1e-12f);
    for (int k = lane; k < CDIM; k += 32) {
        float y = x[k] * inv;
        size_t o = (size_t)row * CDIM + k;
        if (outF) outF[o] = y;
        u16 h = f2bf(y);
        outH[o] = h;
        outL[o] = f2bf(y - bf2f(h));
    }
}

// ---------------------------------------------------------------------------
// Kernel 2: per-class mean of normalized support rows -> proto_raw[64][1024].
// Each block owns 128 columns; acc[class][col] touched by exactly one thread.
// ---------------------------------------------------------------------------
__global__ __launch_bounds__(128) void proto_accum_kernel(
    const float* __restrict__ sN, const int* __restrict__ labels,
    float* __restrict__ proto_raw)
{
    __shared__ float acc[WAYC][128];
    __shared__ float cnt[WAYC];
    int c  = threadIdx.x;
    int c0 = blockIdx.x * 128;

    for (int wcl = 0; wcl < WAYC; ++wcl) acc[wcl][c] = 0.0f;
    if (c < WAYC) cnt[c] = 0.0f;
    __syncthreads();

    for (int r = 0; r < S_ROWS; ++r) {
        int lab = labels[r];
        acc[lab][c] += sN[(size_t)r * CDIM + c0 + c];
    }
    if (c < WAYC) {
        float n = 0.0f;
        for (int r = 0; r < S_ROWS; ++r) n += (labels[r] == c) ? 1.0f : 0.0f;
        cnt[c] = n;
    }
    __syncthreads();

    for (int wcl = 0; wcl < WAYC; ++wcl) {
        float d = fmaxf(cnt[wcl], 1.0f);
        proto_raw[(size_t)wcl * CDIM + c0 + c] = acc[wcl][c] / d;
    }
}

// ---------------------------------------------------------------------------
// WMMA fragment loaders (this code's contiguous K order, same for A and B; wave32).
// A (16-bit, 16x32):  lanes 0-15: M=lane,    K=0..15 contiguous
//                     lanes 16-31: M=lane-16, K=16..31 contiguous
// B (16-bit, 32x16):  lane&15 = N, lanes<16 hold K=0..15, lanes>=16 K=16..31
//                     -> 16 contiguous bf16 per lane in [N][K] LDS layout.
// ---------------------------------------------------------------------------
union FragA { v16bf v; unsigned u[8]; };
union FragB { v16bf v; uint4 q[2]; };

static __device__ __forceinline__ v16bf load_fragA(const u16* rowp, int kbase) {
    const unsigned* ar = (const unsigned*)rowp; // 16 dwords per 32-elem row
    FragA f;
#pragma unroll
    for (int j = 0; j < 8; ++j) {
        int K = 2 * kbase + 2 * j;
        f.u[j] = ar[K >> 1];
    }
    return f.v;
}
static __device__ __forceinline__ v16bf load_fragB(const u16* p16) {
    FragB f;
    const uint4* bp = (const uint4*)p16;
    f.q[0] = bp[0]; f.q[1] = bp[1];
    return f.v;
}

// ---------------------------------------------------------------------------
// Kernel 3: fused sim GEMM (bf16x3) + per-row top-5 + distances + kNN vote.
// Block: 256 thr (8 waves); wave w owns 16 query rows; 64 blocks cover Q.
// 32 groups of 128 supports; K chunks of 32 double-buffered via async LDS DMA.
// The top-k transpose buffer aliases staging buffer 1 (dead at that point).
// ---------------------------------------------------------------------------
__global__ __launch_bounds__(256) void sim_topk_kernel(
    const u16* __restrict__ qH, const u16* __restrict__ qL,
    const u16* __restrict__ sH, const u16* __restrict__ sL,
    const int* __restrict__ labels,
    float* __restrict__ out_idx, float* __restrict__ out_dist,
    float* __restrict__ out_pred)
{
    // [buffer][array: aH,aL,bH,bL][row][k]  -> 2 * 4 * 8KB = 64KB
    __shared__ alignas(16) u16 smem[2][4][128][32];
    float* simlds = (float*)&smem[1][0][0][0];   // 8*16*17 floats = 8.5KB alias
#define SIMBUF(W, R, C) simlds[((W) * 16 + (R)) * 17 + (C)]

    int tid   = threadIdx.x;
    int lane  = tid & 31;
    int w     = tid >> 5;
    int qbase = blockIdx.x * 128;

    int srow  = tid >> 1;          // staging row 0..127
    int shalf = tid & 1;           // staging K-half (16 elems = 32B)

    int myArow = w * 16 + (lane & 15);
    int kbase  = (lane < 16) ? 0 : 8;   // A K-phase
    int koff   = (lane < 16) ? 0 : 16;  // B K-half
    int ncol   = lane & 15;
    int rowoff = (lane < 16) ? 0 : 8;   // C M-offset

    float tv[KNN]; int ti[KNN];
#pragma unroll
    for (int j = 0; j < KNN; ++j) { tv[j] = -3.0e38f; ti[j] = 0; }

    for (int g = 0; g < 32; ++g) {
        int sbase = g * 128;
        v8f acc[8];
#pragma unroll
        for (int t = 0; t < 8; ++t)
#pragma unroll
            for (int r = 0; r < 8; ++r) acc[t][r] = 0.0f;

        // ---- prologue: stage k-chunk 0 into buffer 0 ----------------------
        {
            size_t qo = (size_t)(qbase + srow) * CDIM + shalf * 16;
            size_t so = (size_t)(sbase + srow) * CDIM + shalf * 16;
            cp32_async(qH + qo, &smem[0][0][srow][shalf * 16]);
            cp32_async(qL + qo, &smem[0][1][srow][shalf * 16]);
            cp32_async(sH + so, &smem[0][2][srow][shalf * 16]);
            cp32_async(sL + so, &smem[0][3][srow][shalf * 16]);
        }
        wait_async();
        __syncthreads();

        for (int kk = 0; kk < 32; ++kk) {
            int cur = kk & 1;
            // ---- prefetch next chunk into the other buffer ----------------
            if (kk < 31) {
                int k1 = (kk + 1) * 32;
                size_t qo = (size_t)(qbase + srow) * CDIM + k1 + shalf * 16;
                size_t so = (size_t)(sbase + srow) * CDIM + k1 + shalf * 16;
                u16 (*nb)[128][32] = smem[cur ^ 1];
                cp32_async(qH + qo, &nb[0][srow][shalf * 16]);
                cp32_async(qL + qo, &nb[1][srow][shalf * 16]);
                cp32_async(sH + so, &nb[2][srow][shalf * 16]);
                cp32_async(sL + so, &nb[3][srow][shalf * 16]);
            }
            // ---- compute on current buffer --------------------------------
            u16 (*cb)[128][32] = smem[cur];
            v16bf fah = load_fragA(&cb[0][myArow][0], kbase);
            v16bf fal = load_fragA(&cb[1][myArow][0], kbase);
#pragma unroll
            for (int t = 0; t < 8; ++t) {
                v16bf fbh = load_fragB(&cb[2][t * 16 + ncol][koff]);
                v16bf fbl = load_fragB(&cb[3][t * 16 + ncol][koff]);
                acc[t] = __builtin_amdgcn_wmma_f32_16x16x32_bf16(
                    false, fah, false, fbh, (short)0, acc[t], false, false);
                acc[t] = __builtin_amdgcn_wmma_f32_16x16x32_bf16(
                    false, fah, false, fbl, (short)0, acc[t], false, false);
                acc[t] = __builtin_amdgcn_wmma_f32_16x16x32_bf16(
                    false, fal, false, fbh, (short)0, acc[t], false, false);
            }
            wait_async();      // prefetch done (overlapped with the WMMAs)
            __syncthreads();   // all waves done reading cur / writing next
        }

        // ---- top-5 update: dump each 16x16 tile via LDS, lanes 0-15 scan --
        // (simlds aliases buffer 1; its last reader was kk=31, barrier above)
#pragma unroll 1
        for (int t = 0; t < 8; ++t) {
#pragma unroll
            for (int r = 0; r < 8; ++r) SIMBUF(w, r + rowoff, ncol) = acc[t][r];
            __syncthreads();
            if (lane < 16) {
#pragma unroll 1
                for (int j = 0; j < 16; ++j) {
                    float v  = SIMBUF(w, lane, j);
                    int  idx = sbase + t * 16 + j;
                    if (v > tv[KNN - 1]) {
#pragma unroll
                        for (int p = 0; p < KNN; ++p) {
                            if (v > tv[p]) {
                                for (int s2 = KNN - 1; s2 > p; --s2) {
                                    tv[s2] = tv[s2 - 1]; ti[s2] = ti[s2 - 1];
                                }
                                tv[p] = v; ti[p] = idx;
                                break;
                            }
                        }
                    }
                }
            }
            __syncthreads();
        }
    }

    // ---- finalize: indices, distances (1 - sim), majority-vote prediction -
    if (lane < 16) {
        int qrow = qbase + w * 16 + lane;
        int lab[KNN];
#pragma unroll
        for (int j = 0; j < KNN; ++j) {
            out_idx [qrow * KNN + j] = (float)ti[j];
            out_dist[qrow * KNN + j] = 1.0f - tv[j];
            lab[j] = labels[ti[j]];
        }
        int bestLab = 1 << 30, bestCnt = -1;
#pragma unroll
        for (int j = 0; j < KNN; ++j) {
            int c = 0;
#pragma unroll
            for (int m = 0; m < KNN; ++m) c += (lab[m] == lab[j]) ? 1 : 0;
            if (c > bestCnt || (c == bestCnt && lab[j] < bestLab)) {
                bestCnt = c; bestLab = lab[j];
            }
        }
        out_pred[qrow] = (float)bestLab;
    }
#undef SIMBUF
}

// ---------------------------------------------------------------------------
// Kernel 4: scores = q . proto^T (M=8192, N=64, K=1024), bf16x3 WMMA,
// same async double-buffered staging. Writes 10*scores and scores.
// ---------------------------------------------------------------------------
__global__ __launch_bounds__(256) void scores_kernel(
    const u16* __restrict__ qH, const u16* __restrict__ qL,
    const u16* __restrict__ pH, const u16* __restrict__ pL,
    float* __restrict__ out_cls, float* __restrict__ out_sc)
{
    __shared__ alignas(16) u16 asm2[2][2][128][32];  // [buf][hi/lo] 32KB
    __shared__ alignas(16) u16 bsm [2][2][64][32];   // [buf][hi/lo] 16KB

    int tid   = threadIdx.x;
    int lane  = tid & 31;
    int w     = tid >> 5;
    int qbase = blockIdx.x * 128;

    int srow  = tid >> 1;
    int shalf = tid & 1;
    int brow  = tid >> 2;          // 0..63
    int bcol  = (tid & 3) * 8;     // 0,8,16,24

    int myArow = w * 16 + (lane & 15);
    int kbase  = (lane < 16) ? 0 : 8;
    int koff   = (lane < 16) ? 0 : 16;
    int ncol   = lane & 15;
    int rowoff = (lane < 16) ? 0 : 8;

    v8f acc[4];
#pragma unroll
    for (int t = 0; t < 4; ++t)
#pragma unroll
        for (int r = 0; r < 8; ++r) acc[t][r] = 0.0f;

    // prologue: stage chunk 0 into buffer 0
    {
        size_t qo = (size_t)(qbase + srow) * CDIM + shalf * 16;
        size_t po = (size_t)brow * CDIM + bcol;
        cp32_async(qH + qo, &asm2[0][0][srow][shalf * 16]);
        cp32_async(qL + qo, &asm2[0][1][srow][shalf * 16]);
        cp16_async(pH + po, &bsm[0][0][brow][bcol]);
        cp16_async(pL + po, &bsm[0][1][brow][bcol]);
    }
    wait_async();
    __syncthreads();

    for (int kk = 0; kk < 32; ++kk) {
        int cur = kk & 1;
        if (kk < 31) {
            int k1 = (kk + 1) * 32;
            size_t qo = (size_t)(qbase + srow) * CDIM + k1 + shalf * 16;
            size_t po = (size_t)brow * CDIM + k1 + bcol;
            cp32_async(qH + qo, &asm2[cur ^ 1][0][srow][shalf * 16]);
            cp32_async(qL + qo, &asm2[cur ^ 1][1][srow][shalf * 16]);
            cp16_async(pH + po, &bsm[cur ^ 1][0][brow][bcol]);
            cp16_async(pL + po, &bsm[cur ^ 1][1][brow][bcol]);
        }
        v16bf fah = load_fragA(&asm2[cur][0][myArow][0], kbase);
        v16bf fal = load_fragA(&asm2[cur][1][myArow][0], kbase);
#pragma unroll
        for (int t = 0; t < 4; ++t) {
            v16bf fbh = load_fragB(&bsm[cur][0][t * 16 + ncol][koff]);
            v16bf fbl = load_fragB(&bsm[cur][1][t * 16 + ncol][koff]);
            acc[t] = __builtin_amdgcn_wmma_f32_16x16x32_bf16(
                false, fah, false, fbh, (short)0, acc[t], false, false);
            acc[t] = __builtin_amdgcn_wmma_f32_16x16x32_bf16(
                false, fah, false, fbl, (short)0, acc[t], false, false);
            acc[t] = __builtin_amdgcn_wmma_f32_16x16x32_bf16(
                false, fal, false, fbh, (short)0, acc[t], false, false);
        }
        wait_async();
        __syncthreads();
    }

#pragma unroll
    for (int t = 0; t < 4; ++t)
#pragma unroll
        for (int r = 0; r < 8; ++r) {
            int m   = qbase + w * 16 + r + rowoff;
            int col = t * 16 + ncol;
            float v = acc[t][r];
            out_cls[(size_t)m * WAYC + col] = 10.0f * v;
            out_sc [(size_t)m * WAYC + col] = v;
        }
}

// ---------------------------------------------------------------------------
// Host-side launch. Inputs (dict order): proto_support_images[4096*1024] f32,
// proto_support_labels[4096] i32, query_images[8192*1024] f32.
// Output (flat f32): cls_scores | indices | distances | predictions | scores.
// ---------------------------------------------------------------------------
extern "C" void kernel_launch(void* const* d_in, const int* in_sizes, int n_in,
                              void* d_out, int out_size, void* d_ws, size_t ws_size,
                              hipStream_t stream)
{
    (void)in_sizes; (void)n_in; (void)out_size; (void)ws_size;

    const float* simg   = (const float*)d_in[0];
    const int*   labels = (const int*)  d_in[1];
    const float* qimg   = (const float*)d_in[2];
    float* out = (float*)d_out;

    // output offsets (elements)
    const size_t CLS_OFF  = 0;
    const size_t IDX_OFF  = (size_t)Q_ROWS * WAYC;              // 524288
    const size_t DIST_OFF = IDX_OFF + (size_t)Q_ROWS * KNN;     // 565248
    const size_t PRED_OFF = DIST_OFF + (size_t)Q_ROWS * KNN;    // 606208
    const size_t SC_OFF   = PRED_OFF + (size_t)Q_ROWS;          // 614400

    // workspace layout
    char* ws = (char*)d_ws;
    size_t o = 0;
    u16*   q_hi = (u16*)(ws + o);   o += (size_t)Q_ROWS * CDIM * 2;
    u16*   q_lo = (u16*)(ws + o);   o += (size_t)Q_ROWS * CDIM * 2;
    u16*   s_hi = (u16*)(ws + o);   o += (size_t)S_ROWS * CDIM * 2;
    u16*   s_lo = (u16*)(ws + o);   o += (size_t)S_ROWS * CDIM * 2;
    float* s_nm = (float*)(ws + o); o += (size_t)S_ROWS * CDIM * 4;
    float* p_rw = (float*)(ws + o); o += (size_t)WAYC * CDIM * 4;
    u16*   p_hi = (u16*)(ws + o);   o += (size_t)WAYC * CDIM * 2;
    u16*   p_lo = (u16*)(ws + o);   o += (size_t)WAYC * CDIM * 2;

    l2norm_split_kernel<<<Q_ROWS / 8, 256, 0, stream>>>(qimg, Q_ROWS, nullptr, q_hi, q_lo);
    l2norm_split_kernel<<<S_ROWS / 8, 256, 0, stream>>>(simg, S_ROWS, s_nm,   s_hi, s_lo);
    proto_accum_kernel <<<CDIM / 128, 128, 0, stream>>>(s_nm, labels, p_rw);
    l2norm_split_kernel<<<WAYC / 8,   256, 0, stream>>>(p_rw, WAYC, nullptr, p_hi, p_lo);
    sim_topk_kernel    <<<Q_ROWS / 128, 256, 0, stream>>>(q_hi, q_lo, s_hi, s_lo, labels,
                                                          out + IDX_OFF, out + DIST_OFF,
                                                          out + PRED_OFF);
    scores_kernel      <<<Q_ROWS / 128, 256, 0, stream>>>(q_hi, q_lo, p_hi, p_lo,
                                                          out + CLS_OFF, out + SC_OFF);
}
